// FasterRCNN_34093450396615
// MI455X (gfx1250) — matrix operand order repaired and checked by reading, not compiled, not run on hardware
//
#include <hip/hip_runtime.h>

// ---------------- types for WMMA (gfx1250, wave32) ----------------
typedef __attribute__((ext_vector_type(16))) __bf16         v16bf;
typedef __attribute__((ext_vector_type(8)))  float          v8f;
typedef __attribute__((ext_vector_type(8)))  unsigned short ush8;

union Frag { struct { ush8 lo, hi; } p; v16bf v; };

__device__ __forceinline__ unsigned short f32_to_bf16(float f) {
    unsigned u = __builtin_bit_cast(unsigned, f);
    u += 0x7FFFu + ((u >> 16) & 1u);          // round-to-nearest-even
    return (unsigned short)(u >> 16);
}

// ---------------- problem constants ----------------
#define BB     4
#define CIN    2048
#define HH     50
#define WW     50
#define HW     (HH * WW)        // 2500
#define PADH   52
#define PADW   52
#define CH     512
#define K9     (CIN * 9)        // 18432; k = r*2048 + c  (r = ky*3+kx)
#define MM     (BB * HH * WW)   // 10000
#define MTILE  128
#define NTILE  64
#define KT     32
#define LDA    40               // padded k-stride (ushorts): 80B rows, 16B chunks
#define NKT    (K9 / KT)        // 576 K-tiles
#define NNT    (CH / 16)        // 32 N-tiles of 16
#define NANCH  22500
#define NPAD   22528
#define NSORT  32768
#define PRE_K  2000
#define TOP_N  1000

// =================================================================
// Kernel P0: zero-fill padded activation tensor (bf16, BHWC, 52x52)
// =================================================================
__global__ __launch_bounds__(256)
void pad_zero(unsigned short* __restrict__ fpad) {
    const size_t n = (size_t)BB * PADH * PADW * CIN;
    const size_t i = ((size_t)blockIdx.x * 256 + threadIdx.x) * 8;
    if (i < n) { ush8 z = {}; *(ush8*)(fpad + i) = z; }
}

// =================================================================
// Kernel P1: NCHW f32 -> padded BHWC bf16 transpose via LDS tile.
// grid (hw-tiles=40, c-tiles=32, B), block 256, tile 64c x 64hw.
// =================================================================
__global__ __launch_bounds__(256)
void pad_transpose(const float* __restrict__ feat, unsigned short* __restrict__ fpad) {
    __shared__ __attribute__((aligned(16))) unsigned short tile[64][72];
    const int b   = blockIdx.z;
    const int c0  = blockIdx.y * 64;
    const int hw0 = blockIdx.x * 64;
    const int t   = threadIdx.x;

    { // read: coalesced along hw, one channel per thread-quad
        const int c_l = t >> 2;
        const int hwc = (t & 3) * 16;
        const float* src = feat + ((long)b * CIN + c0 + c_l) * HW + hw0 + hwc;
        unsigned short tmp[16];
        #pragma unroll
        for (int e = 0; e < 16; ++e)
            tmp[e] = f32_to_bf16((hw0 + hwc + e < HW) ? src[e] : 0.0f);
        *(ush8*)&tile[c_l][hwc]     = *(ush8*)&tmp[0];
        *(ush8*)&tile[c_l][hwc + 8] = *(ush8*)&tmp[8];
    }
    __syncthreads();
    { // write: coalesced along c (channel-innermost destination)
        const int hw_l = t >> 2;
        const int cc   = (t & 3) * 16;
        const int hw   = hw0 + hw_l;
        if (hw < HW) {
            unsigned short tmp[16];
            #pragma unroll
            for (int e = 0; e < 16; ++e) tmp[e] = tile[cc + e][hw_l];
            const int y = hw / WW, x = hw - y * WW;
            unsigned short* dst = fpad
                + ((((long)b * PADH + y + 1) * PADW) + (x + 1)) * CIN + c0 + cc;
            *(ush8*)(dst)     = *(ush8*)&tmp[0];
            *(ush8*)(dst + 8) = *(ush8*)&tmp[8];
        }
    }
}

// =================================================================
// Kernel P2: pack weights into WMMA B-fragment layout (bf16).
// wpack[((kt*NNT + nt)*32 + lane)*16 + e]
// =================================================================
__global__ __launch_bounds__(256)
void pack_weights(const float* __restrict__ wconv, unsigned short* __restrict__ wpack) {
    const int gw   = blockIdx.x * 8 + (threadIdx.x >> 5);
    const int lane = threadIdx.x & 31;
    if (gw >= NKT * NNT) return;
    const int kt  = gw >> 5;
    const int nt  = gw & (NNT - 1);
    const int o   = nt * 16 + (lane & 15);
    const int kkb = (lane < 16) ? 0 : 16;

    unsigned short tmp[16];
    #pragma unroll
    for (int e = 0; e < 16; ++e) {
        const int kp = kt * KT + kkb + e;   // k' = r*2048 + c
        const int r  = kp >> 11;
        const int c  = kp & 2047;
        tmp[e] = f32_to_bf16(wconv[(long)o * K9 + c * 9 + r]);
    }
    ush8* dst = (ush8*)(wpack + ((size_t)gw * 32 + lane) * 16);
    dst[0] = *(ush8*)&tmp[0];
    dst[1] = *(ush8*)&tmp[8];
}

// =================================================================
// Kernel 1: 3x3 conv implicit GEMM, bf16 WMMA, f32 accum, ReLU.
// Padded BHWC bf16 activations -> A staging is 2 unconditional
// global_load_b128 + 2 ds_store_b128; double-buffered LDS, one
// barrier per K-step; B fragments from packed global (L2-resident).
// =================================================================
__global__ __launch_bounds__(256)
void conv3x3_wmma(const unsigned short* __restrict__ fpad,
                  const unsigned short* __restrict__ wpack,
                  const float* __restrict__ bconv,
                  float* __restrict__ hout) {
    __shared__ __attribute__((aligned(16))) unsigned short ldsA[2][MTILE * LDA];

    const int tid    = threadIdx.x;
    const int wave   = tid >> 5;
    const int lane   = tid & 31;
    const int m_base = blockIdx.x * MTILE;
    const int n_base = blockIdx.y * NTILE;

    // A-staging coords: thread owns half a row (16 channels per step)
    const int arow = tid >> 1;          // 0..127
    const int ach  = (tid & 1) * 16;    // 0 / 16
    const int am   = m_base + arow;
    const int am_e = (am < MM) ? am : (MM - 1);   // clamp tail rows (stores guarded)
    const int ab   = am_e / HW;
    const int arem = am_e % HW;
    const int ay   = arem / WW, ax = arem % WW;
    const unsigned short* abase = fpad
        + ((((long)ab * PADH + ay + 1) * PADW) + (ax + 1)) * CIN + ach;

    // Fragment read coords (fixed)
    const int arowl = wave * 16 + (lane & 15);
    const int kb    = (lane < 16) ? 0 : 16;  // A: K{0..15} / K{16..31}
    const int nt0   = n_base >> 4;
    const unsigned short* wbase = wpack + ((size_t)nt0 * 32 + lane) * 16;

    v8f acc0 = {}, acc1 = {}, acc2 = {}, acc3 = {};
    int p = 0;

    for (int r = 0; r < 9; ++r) {
        const int tap = ((r / 3 - 1) * PADW + (r % 3 - 1)) * CIN;
        const unsigned short* src = abase + tap;
        ush8 ra = *(const ush8*)(src);
        ush8 rb = *(const ush8*)(src + 8);

        for (int c0 = 0; c0 < CIN; c0 += KT) {
            unsigned short* st = &ldsA[p][arow * LDA + ach];
            *(ush8*)(st)     = ra;
            *(ush8*)(st + 8) = rb;
            if (c0 + KT < CIN) {           // prefetch next step (overlaps WMMA)
                ra = *(const ush8*)(src + c0 + KT);
                rb = *(const ush8*)(src + c0 + KT + 8);
            }
            __syncthreads();

            Frag fa;
            fa.p.lo = *(const ush8*)&ldsA[p][arowl * LDA + kb];
            fa.p.hi = *(const ush8*)&ldsA[p][arowl * LDA + kb + 8];

            const ush8* wp = (const ush8*)wbase;
            Frag fb0, fb1, fb2, fb3;
            fb0.p.lo = wp[0];       fb0.p.hi = wp[1];
            fb1.p.lo = wp[64];      fb1.p.hi = wp[65];    // +1 n-tile = 512 ush
            fb2.p.lo = wp[128];     fb2.p.hi = wp[129];
            fb3.p.lo = wp[192];     fb3.p.hi = wp[193];

            acc0 = __builtin_amdgcn_wmma_f32_16x16x32_bf16(false, fa.v, false, fb0.v, (short)0, acc0, false, false);
            acc1 = __builtin_amdgcn_wmma_f32_16x16x32_bf16(false, fa.v, false, fb1.v, (short)0, acc1, false, false);
            acc2 = __builtin_amdgcn_wmma_f32_16x16x32_bf16(false, fa.v, false, fb2.v, (short)0, acc2, false, false);
            acc3 = __builtin_amdgcn_wmma_f32_16x16x32_bf16(false, fa.v, false, fb3.v, (short)0, acc3, false, false);

            wbase += 16384;   // next k-tile: NNT*32*16 ush
            p ^= 1;
        }
    }

    // ---- store C/D: lane<16 -> N=lane, M=v ; lane>=16 -> N=lane-16, M=v+8
    const int mrow0 = m_base + wave * 16 + ((lane >= 16) ? 8 : 0);
    const int ncol  = n_base + (lane & 15);
    const float bia0 = bconv[ncol];
    const float bia1 = bconv[ncol + 16];
    const float bia2 = bconv[ncol + 32];
    const float bia3 = bconv[ncol + 48];
    #pragma unroll
    for (int v = 0; v < 8; ++v) {
        const int m = mrow0 + v;
        if (m < MM) {
            float* dst = hout + (long)m * CH + ncol;
            dst[0]  = fmaxf(acc0[v] + bia0, 0.0f);
            dst[16] = fmaxf(acc1[v] + bia1, 0.0f);
            dst[32] = fmaxf(acc2[v] + bia2, 0.0f);
            dst[48] = fmaxf(acc3[v] + bia3, 0.0f);
        }
    }
}

// =================================================================
// Kernel 2: 1x1 heads + softmax-fg + anchor decode + clip
// =================================================================
__global__ __launch_bounds__(64)
void rpn_heads(const float* __restrict__ h,
               const float* __restrict__ wcls, const float* __restrict__ bcls,
               const float* __restrict__ wreg, const float* __restrict__ breg,
               const int* __restrict__ imgh_p, const int* __restrict__ imgw_p,
               float* __restrict__ scores, float* __restrict__ boxes) {
    __shared__ float hs[CH];
    __shared__ float outs[64];
    const int m = blockIdx.x;
    const int t = threadIdx.x;

    for (int i = t; i < CH; i += 64) hs[i] = h[(long)m * CH + i];
    __syncthreads();

    if (t < 54) {
        const float* w = (t < 18) ? (wcls + t * CH) : (wreg + (t - 18) * CH);
        float s = (t < 18) ? bcls[t] : breg[t - 18];
        #pragma unroll 8
        for (int c = 0; c < CH; ++c) s += hs[c] * w[c];
        outs[t] = s;
    }
    __syncthreads();

    if (t < 9) {
        const int a = t;
        const int b = m / HW;
        const int rr = m % HW;
        const int y = rr / WW, x = rr % WW;

        const float cls0 = outs[a], cls1 = outs[9 + a];
        const float score = 1.0f / (1.0f + __expf(cls0 - cls1)); // softmax fg

        const float dx = outs[18 + a * 4 + 0];
        const float dy = outs[18 + a * 4 + 1];
        const float dw = outs[18 + a * 4 + 2];
        const float dh = outs[18 + a * 4 + 3];

        const float sizes[3] = {128.0f, 256.0f, 512.0f};
        const float sqr[3]   = {0.70710678f, 1.0f, 1.41421356f}; // sqrt(ratio)
        const float sz = sizes[a / 3], q = sqr[a % 3];
        const float aw = sz / q, ah = sz * q;
        const float acx = (x + 0.5f) * 32.0f, acy = (y + 0.5f) * 32.0f;

        const float pcx = dx * aw + acx;
        const float pcy = dy * ah + acy;
        const float pw  = __expf(fminf(dw, 4.0f)) * aw;
        const float ph  = __expf(fminf(dh, 4.0f)) * ah;

        const float iw = (float)imgw_p[0], ih = (float)imgh_p[0];
        const float x0 = fminf(fmaxf(pcx - 0.5f * pw, 0.0f), iw);
        const float y0 = fminf(fmaxf(pcy - 0.5f * ph, 0.0f), ih);
        const float x1 = fminf(fmaxf(pcx + 0.5f * pw, 0.0f), iw);
        const float y1 = fminf(fmaxf(pcy + 0.5f * ph, 0.0f), ih);

        const int n = (y * WW + x) * 9 + a;
        scores[(long)b * NPAD + n] = score;
        float* bb = boxes + ((long)b * NPAD + n) * 4;
        bb[0] = x0; bb[1] = y0; bb[2] = x1; bb[3] = y1;
    }
}

// =================================================================
// Kernel 3: per-batch full bitonic sort (desc) of 32768 in 256KB LDS
// (CDNA5: 320KB LDS per WGP) — grid 4, block 1024, dyn LDS 256KB
// =================================================================
__global__ __launch_bounds__(1024)
void sort_topk(const float* __restrict__ scores, const float* __restrict__ boxes,
               float* __restrict__ top_boxes, float* __restrict__ top_scores) {
    extern __shared__ __attribute__((aligned(16))) unsigned char smem[];
    float* key = (float*)smem;
    int*   idx = (int*)(smem + (size_t)NSORT * sizeof(float));
    const int b = blockIdx.x, t = threadIdx.x;

    for (int i = t; i < NSORT; i += 1024) {
        key[i] = (i < NANCH) ? scores[(long)b * NPAD + i] : -1e30f;
        idx[i] = i;
    }
    __syncthreads();

    for (int k = 2; k <= NSORT; k <<= 1) {
        for (int j = k >> 1; j > 0; j >>= 1) {
            for (int i = t; i < NSORT; i += 1024) {
                const int ixj = i ^ j;
                if (ixj > i) {
                    const bool desc = ((i & k) == 0);  // overall descending
                    const float ki = key[i], kj = key[ixj];
                    const bool sw = desc ? (ki < kj) : (ki > kj);
                    if (sw) {
                        key[i] = kj; key[ixj] = ki;
                        const int ti = idx[i]; idx[i] = idx[ixj]; idx[ixj] = ti;
                    }
                }
            }
            __syncthreads();
        }
    }

    for (int i = t; i < PRE_K; i += 1024) {
        int s = idx[i]; if (s > NANCH - 1) s = NANCH - 1;
        top_scores[(long)b * 2048 + i] = key[i];
        const float* src = boxes + ((long)b * NPAD + s) * 4;
        float* dst = top_boxes + ((long)b * 2048 + i) * 4;
        dst[0] = src[0]; dst[1] = src[1]; dst[2] = src[2]; dst[3] = src[3];
    }
}

// =================================================================
// Kernel 4: suppression bitmask  masks[b][i][w] : bit j set if j>i & IoU>0.7
// =================================================================
__global__ __launch_bounds__(64)
void nms_iou(const float* __restrict__ top_boxes, unsigned* __restrict__ masks) {
    const int i = blockIdx.x;
    const int b = blockIdx.y;
    const int w = threadIdx.x;
    if (w >= 63) return;

    const float* bi = top_boxes + ((long)b * 2048 + i) * 4;
    const float x0 = bi[0], y0 = bi[1], x1 = bi[2], y1 = bi[3];
    const float area_i = (x1 - x0) * (y1 - y0);

    unsigned bits = 0;
    #pragma unroll 4
    for (int jj = 0; jj < 32; ++jj) {
        const int j = w * 32 + jj;
        if (j > i && j < PRE_K) {
            const float* bj = top_boxes + ((long)b * 2048 + j) * 4;
            const float xx0 = fmaxf(x0, bj[0]), yy0 = fmaxf(y0, bj[1]);
            const float xx1 = fminf(x1, bj[2]), yy1 = fminf(y1, bj[3]);
            const float iw  = fmaxf(xx1 - xx0, 0.0f);
            const float ihh = fmaxf(yy1 - yy0, 0.0f);
            const float inter = iw * ihh;
            const float area_j = (bj[2] - bj[0]) * (bj[3] - bj[1]);
            const float iou = inter / (area_i + area_j - inter + 1e-9f);
            if (iou > 0.7f) bits |= (1u << jj);
        }
    }
    masks[((long)b * PRE_K + i) * 64 + w] = bits;
}

// =================================================================
// Kernel 5: greedy scan, one wave32 per batch; suppression state = 63
// words spread over lanes, broadcast via __shfl (warpSize==32).
// =================================================================
__global__ __launch_bounds__(32)
void nms_scan(const unsigned* __restrict__ masks,
              const float* __restrict__ top_boxes,
              float* __restrict__ out) {
    const int b = blockIdx.x;
    const int lane = threadIdx.x;
    unsigned sup0 = 0, sup1 = 0, keep0 = 0, keep1 = 0;
    int cnt = 0;

    for (int i = 0; i < PRE_K; ++i) {
        const int w = i >> 5, bit = i & 31;
        const unsigned word = (w < 32) ? __shfl(sup0, w, 32) : __shfl(sup1, w - 32, 32);
        const bool suppr = (word >> bit) & 1u;
        if (!suppr) {  // uniform across wave
            if (w < 32) { if (lane == w) keep0 |= (1u << bit); }
            else        { if (lane == w - 32) keep1 |= (1u << bit); }
            const unsigned* mi = masks + ((long)b * PRE_K + i) * 64;
            sup0 |= mi[lane];
            if (lane < 31) sup1 |= mi[lane + 32];
            if (cnt < TOP_N && lane < 4)
                out[((long)b * TOP_N + cnt) * 4 + lane] =
                    top_boxes[((long)b * 2048 + i) * 4 + lane];
            cnt++;
        }
    }
    // pad with suppressed boxes in index order (top_k tie behavior)
    for (int i = 0; i < PRE_K && cnt < TOP_N; ++i) {
        const int w = i >> 5, bit = i & 31;
        const unsigned kw = (w < 32) ? __shfl(keep0, w, 32) : __shfl(keep1, w - 32, 32);
        if (!((kw >> bit) & 1u)) {
            if (lane < 4)
                out[((long)b * TOP_N + cnt) * 4 + lane] =
                    top_boxes[((long)b * 2048 + i) * 4 + lane];
            cnt++;
        }
    }
}

// =================================================================
extern "C" void kernel_launch(void* const* d_in, const int* in_sizes, int n_in,
                              void* d_out, int out_size, void* d_ws, size_t ws_size,
                              hipStream_t stream) {
    const float* feat   = (const float*)d_in[0];
    const float* w_conv = (const float*)d_in[1];
    const float* b_conv = (const float*)d_in[2];
    const float* w_cls  = (const float*)d_in[3];
    const float* b_cls  = (const float*)d_in[4];
    const float* w_reg  = (const float*)d_in[5];
    const float* b_reg  = (const float*)d_in[6];
    const int*   img_h  = (const int*)d_in[7];
    const int*   img_w  = (const int*)d_in[8];
    float* out = (float*)d_out;

    char* ws = (char*)d_ws;
    size_t o = 0;
    float* h             = (float*)(ws + o);          o += (size_t)10112 * CH * 4;
    float* scores        = (float*)(ws + o);          o += (size_t)BB * NPAD * 4;
    float* boxes         = (float*)(ws + o);          o += (size_t)BB * NPAD * 16;
    float* top_boxes     = (float*)(ws + o);          o += (size_t)BB * 2048 * 16;
    float* top_scores    = (float*)(ws + o);          o += (size_t)BB * 2048 * 4;
    unsigned* masks      = (unsigned*)(ws + o);       o += (size_t)BB * PRE_K * 64 * 4;
    unsigned short* wpk  = (unsigned short*)(ws + o); o += (size_t)NKT * NNT * 512 * 2;
    unsigned short* fpad = (unsigned short*)(ws + o); // 4*52*52*2048 bf16 = 44.3MB

    const size_t npad_elems = (size_t)BB * PADH * PADW * CIN;
    pad_zero<<<(unsigned)(npad_elems / (256 * 8)), 256, 0, stream>>>(fpad);

    dim3 gt((HW + 63) / 64, CIN / 64, BB);              // (40, 32, 4)
    pad_transpose<<<gt, 256, 0, stream>>>(feat, fpad);

    pack_weights<<<(NKT * NNT + 7) / 8, 256, 0, stream>>>(w_conv, wpk);

    dim3 g1((MM + MTILE - 1) / MTILE, CH / NTILE);      // (79, 8)
    conv3x3_wmma<<<g1, 256, 0, stream>>>(fpad, wpk, b_conv, h);

    rpn_heads<<<MM, 64, 0, stream>>>(h, w_cls, b_cls, w_reg, b_reg,
                                     img_h, img_w, scores, boxes);

    sort_topk<<<BB, 1024, (size_t)NSORT * 8, stream>>>(scores, boxes,
                                                       top_boxes, top_scores);

    dim3 g4(PRE_K, BB);
    nms_iou<<<g4, 64, 0, stream>>>(top_boxes, masks);

    nms_scan<<<BB, 32, 0, stream>>>(masks, top_boxes, out);
}
